// STR_Loss_38680475468306
// MI455X (gfx1250) — matrix operand order repaired and checked by reading, not compiled, not run on hardware
//
#include <hip/hip_runtime.h>
#include <hip/hip_bf16.h>
#include <math.h>

typedef __attribute__((ext_vector_type(16))) _Float16 v16h;
typedef __attribute__((ext_vector_type(8)))  float    v8f;

#define NB   512
#define HH   32
#define WW   128
#define LL   64
#define TMX  26

// accumulator slots in workspace
#define A_SEG      0
#define A_IRR      1
#define A_CCE      2
#define A_CEL_MID  3
#define A_DICE_MID 4
#define A_CEL_LOW  5
#define A_DICE_LOW 6

__device__ __forceinline__ float block_sum(float v) {
  __shared__ float sm[256];
  int tid = threadIdx.x;
  sm[tid] = v;
  __syncthreads();
  for (int s = blockDim.x >> 1; s > 0; s >>= 1) {
    if (tid < s) sm[tid] += sm[tid + s];
    __syncthreads();
  }
  float r = sm[0];
  __syncthreads();
  return r;
}

__global__ void k_init(float* acc) {
  if (threadIdx.x < 8) acc[threadIdx.x] = 0.f;
}

// ---- SegLoss (CE of softmax probs used as logits) + fore mask, one pass over backfore ----
__global__ void k_seg_fore(const float* __restrict__ masks, const float* __restrict__ bf,
                           float* __restrict__ fore, float* __restrict__ acc) {
  int idx = blockIdx.x * 256 + threadIdx.x;           // exactly NB*HH*WW threads
  int b = idx >> 12; int p = idx & 4095;
  const float* bb = bf + b * 2 * 4096;
  float z0 = bb[p], z1 = bb[p + 4096];
  float p1 = 1.f / (1.f + expf(z0 - z1));             // softmax channel 1
  float p0 = 1.f - p1;
  int   label = masks[idx] > 0.5f;
  float ce = label ? log1pf(expf(p0 - p1)) : log1pf(expf(p1 - p0));
  fore[idx] = (z1 > z0) ? 1.f : 0.f;                  // sm1 > 0.5
  float s = block_sum(ce);
  if (threadIdx.x == 0) atomicAdd(&acc[A_SEG], s);
}

// ---- interp[b] = Attn[b] (26x64) x R (64x128) via WMMA f16 ----
// Two OVERLAPPING 16-row M-tiles per sample: rows [0..15] and [10..25].
// Rows 10..15 are computed twice from identical inputs (benign duplicate store),
// which makes every A load and every D store unconditional: zero exec-mask traffic.
// One wave per M-tile; A fragments loaded once, B fragments VALU-generated per N-tile.
__global__ void k_interp(const float* __restrict__ seq, float* __restrict__ interp) {
  int blk = blockIdx.x;
  int b    = blk >> 1;
  int base = (blk & 1) * 10;                   // M-tile row base: 0 or 10
  int lane = threadIdx.x;
  int Ml = lane & 15;
  int hi = lane >> 4;
  int t  = base + Ml;                          // A row, always < 26
  const float* A = seq + b * LL * TMX;         // seq[b, l, t]; A[t][k] = seq[b][k][t]

  v16h a[2];
#pragma unroll
  for (int s = 0; s < 2; ++s) {
#pragma unroll
    for (int e = 0; e < 16; ++e) {
      // ISA 16-bit A 16x32 layout: K = s*32 + (e>>3)*16 + hi*8 + (e&7)
      int K = s * 32 + ((e >> 3) << 4) + (hi << 3) + (e & 7);
      a[s][e] = (_Float16)A[K * TMX + t];
    }
  }

  for (int nt = 0; nt < 8; ++nt) {
    int wN = nt * 16 + Ml;                     // B/D column for this tile
    float xf = 0.5f * (float)wN - 0.25f;       // half-pixel 64->128 source coord
    int   k0 = (int)floorf(xf);
    float f  = xf - (float)k0;
    int k0c = k0 < 0 ? 0 : k0;
    int k1c = (k0 + 1) > 63 ? 63 : (k0 + 1);
    v8f c = {};
#pragma unroll
    for (int s = 0; s < 2; ++s) {
      v16h bb;
#pragma unroll
      for (int e = 0; e < 16; ++e) {
        int Kb = s * 32 + ((e >> 3) << 4) + (hi << 3) + (e & 7);       // ISA 16-bit B 32x16 layout
        float wt = 0.f;
        if (Kb == k0c) wt += 1.0f - f;
        if (Kb == k1c) wt += f;
        bb[e] = (_Float16)wt;
      }
      c = __builtin_amdgcn_wmma_f32_16x16x32_f16(false, a[s], false, bb, (short)0, c, false, false);
    }
#pragma unroll
    for (int r = 0; r < 8; ++r) {
      int to = base + r + hi * 8;              // always < 26: unconditional store
      interp[(b * TMX + to) * WW + wN] = c[r];
    }
  }
}

// ---- Correct_1D_Location: irrelevant term + char_sum CE, one block per sample ----
__global__ void k_correct(const float* __restrict__ interp, const float* __restrict__ fore,
                          const int* __restrict__ length, float* __restrict__ acc) {
  int b = blockIdx.x;
  int w = threadIdx.x;                         // 128 threads, one per column
  int n = length[b] - 1;
  float nf = (float)n;
  float colsum = 0.f, S = 0.f, S2 = 0.f;
  for (int t = 0; t < TMX; ++t) {
    float v = interp[(b * TMX + t) * WW + w];
    if (t < n) {
      float sq = v * v;
      S += sq; S2 += sq * sq;
      colsum += 1.f / (1.f + expf(-70.f * (v - 0.1f)));
    }
  }
  float irr = 0.5f * (S * S - S2);
  float cs1 = fminf(colsum, 1.f);              // act>=0 so clip(.,0,1)=min(.,1)
  float cesum = 0.f;
  const float* fb = fore + b * 4096;
  for (int h = 0; h < HH; ++h) {
    float f = fb[h * WW + w];
    float cs = (f > 0.5f) ? cs1 : 0.f;         // char_sum = clip(colsum*fore)
    // logits [1-cs, cs], label = fore
    float ce = (f > 0.5f) ? log1pf(expf(1.f - 2.f * cs)) : log1pf(expf(2.f * cs - 1.f));
    cesum += ce;
  }
  float si = block_sum(irr);
  float sc = block_sum(cesum);
  if (threadIdx.x == 0) { atomicAdd(&acc[A_IRR], si / nf); atomicAdd(&acc[A_CCE], sc); }
}

// ---- Char branch: fused fm/ab/label + CE + softmax + mask27 write (low only) ----
template<int HC, int WC, int SC, bool WM>
__global__ void k_char(const float* __restrict__ charf, const float* __restrict__ seq,
                       const float* __restrict__ fore, const int* __restrict__ length,
                       const float* __restrict__ alpha_p,
                       float* __restrict__ soft_out, float* __restrict__ mask_out,
                       float* __restrict__ acc_cel) {
  const int HW = HC * WC;
  int idx = blockIdx.x * 256 + threadIdx.x;
  float ce = 0.f;
  if (idx < NB * HW) {
    float alpha = alpha_p[0];
    int b = idx / HW;
    int p = idx - b * HW;
    int y = p / WC, x = p - y * WC;
    int n = length[b] - 1;
    // fm: half-pixel bilinear downsample of fore, then >= 0.4
    const float* fb = fore + b * 4096;
    float s;
    if (SC == 2) {
      int yy = 2 * y, xx = 2 * x;
      s = fb[yy * WW + xx] + fb[yy * WW + xx + 1] + fb[(yy + 1) * WW + xx] + fb[(yy + 1) * WW + xx + 1];
    } else {
      int yy = 4 * y + 1, xx = 4 * x + 1;
      s = fb[yy * WW + xx] + fb[yy * WW + xx + 1] + fb[(yy + 1) * WW + xx] + fb[(yy + 1) * WW + xx + 1];
    }
    int fm = (0.25f * s >= 0.4f) ? 1 : 0;
    // ab bitmask over 26 chars (attn detached, row_valid applied)
    unsigned abm = 0;
    const float* sb = seq + b * LL * TMX;
    for (int tt = 0; tt < TMX; ++tt) {
      if (tt < n) {
        float av;
        if (SC == 2) av = sb[x * TMX + tt];                                     // 64->64 identity
        else         av = 0.5f * (sb[(2 * x) * TMX + tt] + sb[(2 * x + 1) * TMX + tt]); // 64->32 avg
        if (av > alpha) abm |= (1u << tt);
      }
    }
    int label = (fm && abm) ? __ffs(abm) : 0;   // argmax of mask27 over channels
    // fused 27-way softmax + CE (channels kept in registers, constant-indexed)
    const float* cf = charf + (b * 27) * HW + p;
    float z[27];
    float m = -3.4e38f, zl = 0.f;
#pragma unroll
    for (int c = 0; c < 27; ++c) {
      float zc = cf[c * HW];
      z[c] = zc;
      m = fmaxf(m, zc);
      if (c == label) zl = zc;
    }
    float se = 0.f;
#pragma unroll
    for (int c = 0; c < 27; ++c) { z[c] = expf(z[c] - m); se += z[c]; }
    ce = logf(se) + m - zl;
    float inv = 1.f / se;
#pragma unroll
    for (int c = 1; c < 27; ++c)
      soft_out[(b * TMX + (c - 1)) * HW + p] = z[c] * inv;
    if (WM) {
      mask_out[(b * 27) * HW + p] = fm ? 0.f : 1.f;
#pragma unroll
      for (int tt = 0; tt < TMX; ++tt)
        mask_out[(b * 27 + tt + 1) * HW + p] = (fm && ((abm >> tt) & 1u)) ? 1.f : 0.f;
    }
  }
  float sce = block_sum(ce);
  if (threadIdx.x == 0) atomicAdd(acc_cel, sce);
}

// ---- Dice: one block per (b, t); re-reads soft output, recomputes m2 on the fly ----
template<int HC, int WC, int SC>
__global__ void k_dice(const float* __restrict__ soft_out, const float* __restrict__ seq,
                       const float* __restrict__ fore, const int* __restrict__ length,
                       const float* __restrict__ alpha_p, float* __restrict__ acc_dice) {
  const int HW = HC * WC;
  int b = blockIdx.x / TMX;
  int t = blockIdx.x - b * TMX;
  int n = length[b] - 1;
  float inter = 0.f, s1 = 0.f, s2 = 0.f;
  if (t < n) {
    float alpha = alpha_p[0];
    const float* fb = fore + b * 4096;
    const float* sb = seq + b * LL * TMX;
    const float* sp = soft_out + (b * TMX + t) * HW;
    for (int p = threadIdx.x; p < HW; p += 256) {
      int y = p / WC, x = p - y * WC;
      float s;
      if (SC == 2) {
        int yy = 2 * y, xx = 2 * x;
        s = fb[yy * WW + xx] + fb[yy * WW + xx + 1] + fb[(yy + 1) * WW + xx] + fb[(yy + 1) * WW + xx + 1];
      } else {
        int yy = 4 * y + 1, xx = 4 * x + 1;
        s = fb[yy * WW + xx] + fb[yy * WW + xx + 1] + fb[(yy + 1) * WW + xx] + fb[(yy + 1) * WW + xx + 1];
      }
      int fm = (0.25f * s >= 0.4f) ? 1 : 0;
      float av;
      if (SC == 2) av = sb[x * TMX + t];
      else         av = 0.5f * (sb[(2 * x) * TMX + t] + sb[(2 * x + 1) * TMX + t]);
      int m2 = (fm && (av > alpha)) ? 1 : 0;
      float sv = sp[p];
      if (m2) inter += sv;
      s1 += sv;
      s2 += (float)m2;
    }
  }
  float I   = block_sum(inter);
  float S1  = block_sum(s1);
  float S2v = block_sum(s2);
  if (threadIdx.x == 0 && t < n) {
    float score = (2.f * I + 1.f) / (S1 + S2v + 1.f);
    atomicAdd(acc_dice, (1.f - score) / (float)n);
  }
}

__global__ void k_finalize(const float* __restrict__ acc, float* __restrict__ out) {
  if (threadIdx.x == 0) {
    float inv_px = 1.f / (float)(NB * HH * WW);
    out[0] = acc[A_SEG] * inv_px;                                   // predmask (iteration<=20000 path)
    out[1] = acc[A_CCE] * inv_px + acc[A_IRR];                      // correct_loss
    out[2] = acc[A_CEL_LOW] / (float)(NB * 8 * 32) + acc[A_DICE_LOW] / (float)NB;   // loss_low
    out[3] = acc[A_CEL_MID] / (float)(NB * 16 * 64) + acc[A_DICE_MID] / (float)NB;  // loss_middle
  }
}

extern "C" void kernel_launch(void* const* d_in, const int* in_sizes, int n_in,
                              void* d_out, int out_size, void* d_ws, size_t ws_size,
                              hipStream_t stream) {
  const float* masks    = (const float*)d_in[0];
  const float* bf       = (const float*)d_in[1];
  const float* seq      = (const float*)d_in[2];
  const float* char_mid = (const float*)d_in[3];
  const float* char_low = (const float*)d_in[4];
  const int*   length   = (const int*)d_in[5];
  const float* alpha_p  = (const float*)d_in[7];

  float* out = (float*)d_out;
  float* mask_low = out + 4;
  float* soft_mid = mask_low + (size_t)NB * 27 * 8 * 32;
  float* soft_low = soft_mid + (size_t)NB * 26 * 16 * 64;

  float* ws     = (float*)d_ws;
  float* acc    = ws;
  float* fore   = ws + 8;                    // NB*32*128 floats
  float* interp = fore + (size_t)NB * 4096;  // NB*26*128 floats

  k_init<<<1, 32, 0, stream>>>(acc);
  k_seg_fore<<<(NB * HH * WW) / 256, 256, 0, stream>>>(masks, bf, fore, acc);
  k_interp<<<NB * 2, 32, 0, stream>>>(seq, interp);
  k_correct<<<NB, 128, 0, stream>>>(interp, fore, length, acc);
  k_char<16, 64, 2, false><<<(NB * 16 * 64) / 256, 256, 0, stream>>>(
      char_mid, seq, fore, length, alpha_p, soft_mid, nullptr, acc + A_CEL_MID);
  k_char<8, 32, 4, true><<<(NB * 8 * 32) / 256, 256, 0, stream>>>(
      char_low, seq, fore, length, alpha_p, soft_low, mask_low, acc + A_CEL_LOW);
  k_dice<16, 64, 2><<<NB * TMX, 256, 0, stream>>>(soft_mid, seq, fore, length, alpha_p, acc + A_DICE_MID);
  k_dice<8, 32, 4><<<NB * TMX, 256, 0, stream>>>(soft_low, seq, fore, length, alpha_p, acc + A_DICE_LOW);
  k_finalize<<<1, 32, 0, stream>>>(acc, out);
}
